// GraphTransformerLayer_1984274890918
// MI455X (gfx1250) — matrix operand order repaired and checked by reading, not compiled, not run on hardware
//
#include <hip/hip_runtime.h>
#include <hip/hip_bf16.h>
#include <math.h>

#define N_NODES 50000
#define C_DIM   128
#define H_HEADS 8
#define D_HEAD  16
#define E_EDGES 600000
#define M_TILES (N_NODES / 16)   // 3125

typedef __attribute__((ext_vector_type(16))) _Float16 v16h;
typedef __attribute__((ext_vector_type(8)))  _Float16 v8h;
typedef __attribute__((ext_vector_type(8)))  float    v8f;

// ---------------- ordered-float encoding for atomic max ----------------
__device__ __forceinline__ unsigned enc_ord(float f) {
  unsigned u = __float_as_uint(f);
  return (u & 0x80000000u) ? ~u : (u | 0x80000000u);
}
__device__ __forceinline__ float dec_ord(unsigned u) {
  return __uint_as_float((u & 0x80000000u) ? (u & 0x7FFFFFFFu) : ~u);
}

// ---------------- weight packing into per-lane WMMA B fragments ----------------
// dst flat index t = ((kt*NT + nt)*32 + lane)*16 + i
// maps W[k, n] with k = kt*32 + (lane<16 ? 0 : 16) + i, n = nt*16 + (lane&15)
__global__ void pack_w_kernel(const float* __restrict__ W, _Float16* __restrict__ dst,
                              int Cin, int Cout) {
  int t = blockIdx.x * blockDim.x + threadIdx.x;
  if (t >= Cin * Cout) return;
  int i    = t & 15;
  int lane = (t >> 4) & 31;
  int rest = t >> 9;
  int NT   = Cout >> 4;
  int nt   = rest % NT;
  int kt   = rest / NT;
  int k = kt * 32 + ((lane < 16) ? 0 : 16) + i;
  int n = nt * 16 + (lane & 15);
  dst[t] = (_Float16)W[k * Cout + n];
}

// packs [Wq | Wk | Wv] as one 128x384 matrix
__global__ void pack_qkv_kernel(const float* __restrict__ Wq, const float* __restrict__ Wk,
                                const float* __restrict__ Wv, _Float16* __restrict__ dst) {
  int t = blockIdx.x * blockDim.x + threadIdx.x;
  if (t >= C_DIM * 3 * C_DIM) return;
  int i    = t & 15;
  int lane = (t >> 4) & 31;
  int rest = t >> 9;
  const int NT = 24;
  int nt = rest % NT;
  int kt = rest / NT;
  int k = kt * 32 + ((lane < 16) ? 0 : 16) + i;
  int n = nt * 16 + (lane & 15);
  const float* W = (n < 128) ? Wq : ((n < 256) ? Wk : Wv);
  dst[t] = (_Float16)W[k * C_DIM + (n & 127)];
}

// ---------------- zero accumulators ----------------
__global__ void zero_kernel(float* __restrict__ attn, unsigned* __restrict__ stats) {
  int t = blockIdx.x * blockDim.x + threadIdx.x;
  if (t < N_NODES * C_DIM) attn[t] = 0.0f;
  if (blockIdx.x == 0) {
    if (threadIdx.x < 8) stats[threadIdx.x] = 0u;                    // max (encoded)
    else if (threadIdx.x < 16) ((float*)stats)[threadIdx.x] = 0.0f;  // sums
  }
}

// ---------------- LayerNorm: one wave32 per 128-wide row ----------------
__global__ void ln_kernel(const float* __restrict__ x, const float* __restrict__ g,
                          const float* __restrict__ b, _Float16* __restrict__ o) {
  int lane = threadIdx.x & 31;
  int row  = blockIdx.x * (blockDim.x >> 5) + (threadIdx.x >> 5);
  const float4 xv = *(const float4*)(x + row * C_DIM + lane * 4);
  float s  = xv.x + xv.y + xv.z + xv.w;
  float s2 = xv.x * xv.x + xv.y * xv.y + xv.z * xv.z + xv.w * xv.w;
#pragma unroll
  for (int off = 16; off > 0; off >>= 1) {
    s  += __shfl_xor(s, off);
    s2 += __shfl_xor(s2, off);
  }
  float mean = s * (1.0f / 128.0f);
  float var  = s2 * (1.0f / 128.0f) - mean * mean;
  float rstd = rsqrtf(var + 1e-5f);
  float xs[4] = {xv.x, xv.y, xv.z, xv.w};
#pragma unroll
  for (int j = 0; j < 4; ++j) {
    int c = lane * 4 + j;
    o[row * C_DIM + c] = (_Float16)((xs[j] - mean) * rstd * g[c] + b[c]);
  }
}

// ---------------- register-blocked WMMA: one m-tile x G n-tiles ----------------
// A: row-major f16, Atile points at row mTile*16.  Bbase already offset to n-group.
// Per k-step: batch-load A (2x b128) + all G B fragments (2G x b128) into distinct
// registers, then issue G independent back-to-back WMMAs so loads drain with
// partial s_wait_loadcnt instead of full stalls.
template <int NT, int G>
__device__ __forceinline__ void gemm_tile_g(const _Float16* __restrict__ Atile, int lda,
                                            const v16h* __restrict__ Bbase, int ksteps,
                                            int lane, v8f cacc[G]) {
  const _Float16* ap = Atile + (lane & 15) * lda + ((lane < 16) ? 0 : 16);
  for (int kt = 0; kt < ksteps; ++kt) {
    const v8h* p = (const v8h*)(ap + kt * 32);
    v8h lo = p[0];   // K = kbase + 0..7
    v8h hi = p[1];   // K = kbase + 8..15
    v16h bfrag[G];
#pragma unroll
    for (int j = 0; j < G; ++j) bfrag[j] = Bbase[kt * (NT * 32) + j * 32 + lane];
    v16h a;
#pragma unroll
    for (int i = 0; i < 8; ++i) { a[i] = lo[i]; a[i + 8] = hi[i]; }
#pragma unroll
    for (int j = 0; j < G; ++j) {
      cacc[j] = __builtin_amdgcn_wmma_f32_16x16x32_f16(false, a, false, bfrag[j],
                                                       (short)0, cacc[j], false, false);
    }
  }
}

// ---------------- fused QKV projection: xn[50000x128] @ W[128x384] + bias ----------------
__global__ void gemm_qkv_kernel(const _Float16* __restrict__ A, const v16h* __restrict__ Bp,
                                const float* __restrict__ bq, const float* __restrict__ bk,
                                const float* __restrict__ bv, float* __restrict__ qkv) {
  constexpr int NT = 24, G = 4, NG = NT / G;  // 6 n-groups
  int gw   = (blockIdx.x * blockDim.x + threadIdx.x) >> 5;
  int lane = threadIdx.x & 31;
  int ng = gw % NG, mt = gw / NG;
  if (mt >= M_TILES) return;
  v8f c[G] = {};
  gemm_tile_g<NT, G>(A + mt * 16 * C_DIM, C_DIM, Bp + ng * G * 32, 4, lane, c);
  int rbase = mt * 16 + ((lane >= 16) ? 8 : 0);
#pragma unroll
  for (int j = 0; j < G; ++j) {
    int col = (ng * G + j) * 16 + (lane & 15);
    float bias = (col < 128) ? bq[col] : ((col < 256) ? bk[col - 128] : bv[col - 256]);
#pragma unroll
    for (int r = 0; r < 8; ++r) qkv[(rbase + r) * 384 + col] = c[j][r] + bias;
  }
}

// ---------------- per-edge scores + global per-head max ----------------
__global__ void scores_kernel(const long long* __restrict__ ei, const float* __restrict__ qkv,
                              float* __restrict__ scr, unsigned* __restrict__ gmax) {
  __shared__ unsigned smax[8];
  int tid = threadIdx.x;
  if (tid < 8) smax[tid] = 0u;
  __syncthreads();
  int t = blockIdx.x * blockDim.x + tid;
  int h = t & 7;
  int e = t >> 3;
  int r = (int)ei[e];
  int c = (int)ei[E_EDGES + e];
  const float4* qp = (const float4*)(qkv + r * 384 + h * 16);
  const float4* kp = (const float4*)(qkv + c * 384 + 128 + h * 16);
  float acc = 0.0f;
#pragma unroll
  for (int j = 0; j < 4; ++j) {
    float4 a = qp[j], bb = kp[j];
    acc += a.x * bb.x + a.y * bb.y + a.z * bb.z + a.w * bb.w;
  }
  float sc = acc * 0.25f;  // 1/sqrt(16)
  scr[t] = sc;
  atomicMax(&smax[h], enc_ord(sc));
  __syncthreads();
  if (tid < 8) atomicMax(&gmax[tid], smax[tid]);
}

// ---------------- exp(score - max) and global per-head sum ----------------
__global__ void expsum_kernel(float* __restrict__ scr, const unsigned* __restrict__ gmax,
                              float* __restrict__ gsum) {
  __shared__ float ssum[8];
  int tid = threadIdx.x;
  if (tid < 8) ssum[tid] = 0.0f;
  __syncthreads();
  int t = blockIdx.x * blockDim.x + tid;
  int h = t & 7;
  float ex = expf(scr[t] - dec_ord(gmax[h]));
  scr[t] = ex;
  atomicAdd(&ssum[h], ex);
  __syncthreads();
  if (tid < 8) atomicAdd(&gsum[tid], ssum[tid]);
}

// ---------------- weighted scatter-add of V into node accumulator ----------------
__global__ void scatter_kernel(const long long* __restrict__ ei, const float* __restrict__ scr,
                               const float* __restrict__ gsum, const float* __restrict__ qkv,
                               float* __restrict__ attn) {
  int t = blockIdx.x * blockDim.x + threadIdx.x;
  int h = t & 7;
  int e = t >> 3;
  float w = scr[t] / gsum[h];
  int r = (int)ei[e];
  int c = (int)ei[E_EDGES + e];
  const float* vp = qkv + c * 384 + 256 + h * 16;
  float* op = attn + r * C_DIM + h * 16;
#pragma unroll
  for (int d = 0; d < 16; ++d) atomicAdd(op + d, w * vp[d]);
}

__global__ void cvt_kernel(const float* __restrict__ a, _Float16* __restrict__ o, int n) {
  int t = blockIdx.x * blockDim.x + threadIdx.x;
  if (t < n) o[t] = (_Float16)a[t];
}

// ---------------- O projection + bias + residual ----------------
__global__ void gemm_o_kernel(const _Float16* __restrict__ A, const v16h* __restrict__ Bp,
                              const float* __restrict__ bo, const float* __restrict__ x,
                              float* __restrict__ out1) {
  constexpr int NT = 8, G = 4, NG = NT / G;  // 2 n-groups
  int gw   = (blockIdx.x * blockDim.x + threadIdx.x) >> 5;
  int lane = threadIdx.x & 31;
  int ng = gw % NG, mt = gw / NG;
  if (mt >= M_TILES) return;
  v8f c[G] = {};
  gemm_tile_g<NT, G>(A + mt * 16 * C_DIM, C_DIM, Bp + ng * G * 32, 4, lane, c);
  int rbase = mt * 16 + ((lane >= 16) ? 8 : 0);
#pragma unroll
  for (int j = 0; j < G; ++j) {
    int col = (ng * G + j) * 16 + (lane & 15);
#pragma unroll
    for (int r = 0; r < 8; ++r) {
      int row = rbase + r;
      out1[row * C_DIM + col] = c[j][r] + bo[col] + x[row * C_DIM + col];
    }
  }
}

// ---------------- FFN1: h @ W1[128x512] + b1, exact GELU ----------------
__global__ void gemm_ffn1_kernel(const _Float16* __restrict__ A, const v16h* __restrict__ Bp,
                                 const float* __restrict__ b1, _Float16* __restrict__ act) {
  constexpr int NT = 32, G = 4, NG = NT / G;  // 8 n-groups
  int gw   = (blockIdx.x * blockDim.x + threadIdx.x) >> 5;
  int lane = threadIdx.x & 31;
  int ng = gw % NG, mt = gw / NG;
  if (mt >= M_TILES) return;
  v8f c[G] = {};
  gemm_tile_g<NT, G>(A + mt * 16 * C_DIM, C_DIM, Bp + ng * G * 32, 4, lane, c);
  int rbase = mt * 16 + ((lane >= 16) ? 8 : 0);
#pragma unroll
  for (int j = 0; j < G; ++j) {
    int col = (ng * G + j) * 16 + (lane & 15);
#pragma unroll
    for (int r = 0; r < 8; ++r) {
      float v = c[j][r] + b1[col];
      float ge = 0.5f * v * (1.0f + erff(v * 0.70710678118654752f));
      act[(rbase + r) * 512 + col] = (_Float16)ge;
    }
  }
}

// ---------------- FFN2: act @ W2[512x128] + b2 + residual ----------------
__global__ void gemm_ffn2_kernel(const _Float16* __restrict__ A, const v16h* __restrict__ Bp,
                                 const float* __restrict__ b2, const float* __restrict__ out1,
                                 float* __restrict__ out) {
  constexpr int NT = 8, G = 4, NG = NT / G;  // 2 n-groups
  int gw   = (blockIdx.x * blockDim.x + threadIdx.x) >> 5;
  int lane = threadIdx.x & 31;
  int ng = gw % NG, mt = gw / NG;
  if (mt >= M_TILES) return;
  v8f c[G] = {};
  gemm_tile_g<NT, G>(A + mt * 16 * 512, 512, Bp + ng * G * 32, 16, lane, c);
  int rbase = mt * 16 + ((lane >= 16) ? 8 : 0);
#pragma unroll
  for (int j = 0; j < G; ++j) {
    int col = (ng * G + j) * 16 + (lane & 15);
#pragma unroll
    for (int r = 0; r < 8; ++r) {
      int row = rbase + r;
      out[row * C_DIM + col] = c[j][r] + b2[col] + out1[row * C_DIM + col];
    }
  }
}

extern "C" void kernel_launch(void* const* d_in, const int* in_sizes, int n_in,
                              void* d_out, int out_size, void* d_ws, size_t ws_size,
                              hipStream_t stream) {
  (void)in_sizes; (void)n_in; (void)out_size; (void)ws_size;
  const float*     x    = (const float*)d_in[0];
  const long long* ei   = (const long long*)d_in[1];
  const float*     ln1g = (const float*)d_in[2];
  const float*     ln1b = (const float*)d_in[3];
  const float*     Wq   = (const float*)d_in[4];
  const float*     bq   = (const float*)d_in[5];
  const float*     Wk   = (const float*)d_in[6];
  const float*     bk   = (const float*)d_in[7];
  const float*     Wv   = (const float*)d_in[8];
  const float*     bv   = (const float*)d_in[9];
  const float*     Wo   = (const float*)d_in[10];
  const float*     bo   = (const float*)d_in[11];
  const float*     ln2g = (const float*)d_in[12];
  const float*     ln2b = (const float*)d_in[13];
  const float*     W1   = (const float*)d_in[14];
  const float*     b1   = (const float*)d_in[15];
  const float*     W2   = (const float*)d_in[16];
  const float*     b2   = (const float*)d_in[17];
  float* out = (float*)d_out;

  char* ws = (char*)d_ws;
  _Float16* WqkvP = (_Float16*)(ws + 0);          //  98304 B
  _Float16* WoP   = (_Float16*)(ws + 98304);      //  32768 B
  _Float16* W1P   = (_Float16*)(ws + 131072);     // 131072 B
  _Float16* W2P   = (_Float16*)(ws + 262144);     // 131072 B
  _Float16* xnH   = (_Float16*)(ws + 393216);     // 12.8 MB
  float*    qkv   = (float*)   (ws + 13193216);   // 76.8 MB
  float*    scr   = (float*)   (ws + 89993216);   // 19.2 MB
  float*    attn  = (float*)   (ws + 109193216);  // 25.6 MB
  _Float16* aoH   = (_Float16*)(ws + 134793216);  // 12.8 MB
  float*    out1  = (float*)   (ws + 147593216);  // 25.6 MB
  _Float16* h2H   = (_Float16*)(ws + 173193216);  // 12.8 MB
  _Float16* actH  = (_Float16*)(ws + 185993216);  // 51.2 MB
  unsigned* stats = (unsigned*)(ws + 237193216);  // [0..7]=max enc, [8..15]=sum f32

  dim3 B(256);
  pack_qkv_kernel<<<192, B, 0, stream>>>(Wq, Wk, Wv, WqkvP);
  pack_w_kernel<<<64,  B, 0, stream>>>(Wo, WoP, 128, 128);
  pack_w_kernel<<<256, B, 0, stream>>>(W1, W1P, 128, 512);
  pack_w_kernel<<<256, B, 0, stream>>>(W2, W2P, 512, 128);
  zero_kernel<<<25000, B, 0, stream>>>(attn, stats);
  ln_kernel<<<12500, 128, 0, stream>>>(x, ln1g, ln1b, xnH);
  // waves needed: 3125 m-tiles * 6 n-groups = 18750 -> 2344 blocks of 8 waves
  gemm_qkv_kernel<<<2344, B, 0, stream>>>(xnH, (const v16h*)WqkvP, bq, bk, bv, qkv);
  scores_kernel<<<18750, B, 0, stream>>>(ei, qkv, scr, stats);
  expsum_kernel<<<18750, B, 0, stream>>>(scr, stats, (float*)stats + 8);
  scatter_kernel<<<18750, B, 0, stream>>>(ei, scr, (const float*)stats + 8, qkv, attn);
  cvt_kernel<<<25000, B, 0, stream>>>(attn, aoH, N_NODES * C_DIM);
  // 3125 * 2 n-groups = 6250 waves -> 782 blocks
  gemm_o_kernel<<<782, B, 0, stream>>>(aoH, (const v16h*)WoP, bo, x, out1);
  ln_kernel<<<12500, 128, 0, stream>>>(out1, ln2g, ln2b, h2H);
  // 3125 * 8 n-groups = 25000 waves -> 3125 blocks (exact)
  gemm_ffn1_kernel<<<3125, B, 0, stream>>>(h2H, (const v16h*)W1P, b1, actH);
  // 3125 * 2 n-groups = 6250 waves -> 782 blocks
  gemm_ffn2_kernel<<<782, B, 0, stream>>>(actH, (const v16h*)W2P, b2, out1, out);
}
